// SCNN_4578435137764
// MI455X (gfx1250) — matrix operand order repaired and checked by reading, not compiled, not run on hardware
//
#include <hip/hip_runtime.h>
#include <hip/hip_bf16.h>

// ---------------- problem constants (match reference) ----------------
#define THRESH 0.5f
#define DECAY  0.2f
#define B_     8192
#define IN_    784
#define H_     1024
#define OUT_   10
#define CORES_ 10
#define L_     256
#define TABLE_ 1808
#define TW_    20
#define KP_    800      // IN padded up to a multiple of 32 (f16 WMMA K)
#define N2P_   16       // OUT padded to one WMMA N tile

typedef _Float16 v16h __attribute__((ext_vector_type(16)));
typedef _Float16 v8h  __attribute__((ext_vector_type(8)));
typedef float    v8f  __attribute__((ext_vector_type(8)));

union AFrag { v16h v; v8h h[2]; };

// ---------------- threefry2x32 (JAX-compatible) ----------------
__host__ __device__ __forceinline__ unsigned tf_rotl(unsigned x, int n) {
  return (x << n) | (x >> (32 - n));
}
__host__ __device__ inline void threefry2x32(unsigned k0, unsigned k1,
                                             unsigned c0, unsigned c1,
                                             unsigned& o0, unsigned& o1) {
  unsigned ks2 = 0x1BD11BDAu ^ k0 ^ k1;
  unsigned x0 = c0 + k0, x1 = c1 + k1;
  const int ra[4] = {13, 15, 26, 6};
  const int rb[4] = {17, 29, 16, 24};
#define TF_R(R) { x0 += x1; x1 = tf_rotl(x1, (R)); x1 ^= x0; }
  for (int i = 0; i < 4; ++i) TF_R(ra[i]);  x0 += k1;  x1 += ks2 + 1u;
  for (int i = 0; i < 4; ++i) TF_R(rb[i]);  x0 += ks2; x1 += k0  + 2u;
  for (int i = 0; i < 4; ++i) TF_R(ra[i]);  x0 += k0;  x1 += k1  + 3u;
  for (int i = 0; i < 4; ++i) TF_R(rb[i]);  x0 += k1;  x1 += ks2 + 4u;
  for (int i = 0; i < 4; ++i) TF_R(ra[i]);  x0 += ks2; x1 += k0  + 5u;
#undef TF_R
  o0 = x0; o1 = x1;
}

// ---------------- block-level 3-way reduction helper ----------------
__device__ __forceinline__ void block_reduce3_atomic(float p1, float p2, float p3,
                                                     float* red, float* acc) {
  const int t = threadIdx.x;
  red[t] = p1; __syncthreads();
  for (int st = 128; st > 0; st >>= 1) { if (t < st) red[t] += red[t + st]; __syncthreads(); }
  if (t == 0) atomicAdd(&acc[0], red[0]);
  __syncthreads();
  red[t] = p2; __syncthreads();
  for (int st = 128; st > 0; st >>= 1) { if (t < st) red[t] += red[t + st]; __syncthreads(); }
  if (t == 0) atomicAdd(&acc[1], red[0]);
  __syncthreads();
  red[t] = p3; __syncthreads();
  for (int st = 128; st > 0; st >>= 1) { if (t < st) red[t] += red[t + st]; __syncthreads(); }
  if (t == 0) atomicAdd(&acc[2], red[0]);
}

// ---------------- utility: zero a float region ----------------
__global__ void snn_zero(float* __restrict__ p, long n) {
  long i = (long)blockIdx.x * blockDim.x + threadIdx.x;
  long stride = (long)gridDim.x * blockDim.x;
  for (; i < n; i += stride) p[i] = 0.0f;
}

// ---------------- A1/A2 lookup-table precompute ----------------
__global__ __launch_bounds__(256) void snn_tables(const float* __restrict__ vals,
                                                  const int* __restrict__ counts,
                                                  float* __restrict__ A1,
                                                  float* __restrict__ A2) {
  __shared__ int smax[256];
  const int i = blockIdx.x;     // table row
  const int t = threadIdx.x;    // L index
  float a1 = 0.0f, a2 = 0.0f;
  for (int c = 0; c < CORES_; ++c) {
    float v = vals[((long)c * TABLE_ + i) * L_ + t];
    smax[t] = (v != 0.0f) ? t : -1;
    __syncthreads();
    for (int s = 128; s > 0; s >>= 1) {
      if (t < s) smax[t] = (smax[t] > smax[t + s]) ? smax[t] : smax[t + s];
      __syncthreads();
    }
    if (t == 0) {
      int cnt = counts[c * TABLE_ + i];
      if (cnt > 0) { a1 += (float)(smax[0] - 1); a2 += (float)cnt; }
    }
    __syncthreads();
  }
  if (t == 0) { A1[i] = a1; A2[i] = a2; }
}

// ---------------- weight packing (f32 -> padded f16) ----------------
__global__ void snn_pack_w1(const float* __restrict__ W1, _Float16* __restrict__ w1h) {
  int idx = blockIdx.x * blockDim.x + threadIdx.x;      // over H_*KP_
  if (idx >= H_ * KP_) return;
  int h = idx / KP_, k = idx % KP_;
  w1h[idx] = (k < IN_) ? (_Float16)W1[h * IN_ + k] : (_Float16)0.0f;
}
__global__ void snn_pack_w2(const float* __restrict__ W2, _Float16* __restrict__ w2h) {
  int idx = blockIdx.x * blockDim.x + threadIdx.x;      // over N2P_*H_
  if (idx >= N2P_ * H_) return;
  int n = idx / H_, k = idx % H_;
  w2h[idx] = (n < OUT_) ? (_Float16)W2[n * H_ + k] : (_Float16)0.0f;
}

// ---------------- Bernoulli spike encode + input accounting ----------------
__global__ __launch_bounds__(256) void snn_spikes(const float* __restrict__ input,
                                                  _Float16* __restrict__ xh,
                                                  const float* __restrict__ A1,
                                                  const float* __restrict__ A2,
                                                  float* __restrict__ acc,
                                                  unsigned kk0, unsigned kk1) {
  __shared__ float red[256];
  const int t = threadIdx.x;
  const long idx = (long)blockIdx.x * 256 + t;          // over B_*KP_ (exact)
  const int k = (int)(idx % KP_);
  const int b = (int)(idx / KP_);
  float s = 0.0f, p1 = 0.0f, p2 = 0.0f;
  if (k < IN_) {
    // jax.random.uniform(key, (B,IN)): element e < n/2 -> o0 of counts (e, e+n/2),
    // element e >= n/2 -> o1 of counts (e-n/2, e)
    const long e = (long)b * IN_ + k;
    const long ntot = (long)B_ * IN_, halfn = ntot / 2;
    unsigned c0, c1, o0, o1; int sel;
    if (e < halfn) { c0 = (unsigned)e; c1 = (unsigned)(e + halfn); sel = 0; }
    else           { c0 = (unsigned)(e - halfn); c1 = (unsigned)e; sel = 1; }
    threefry2x32(kk0, kk1, c0, c1, o0, o1);
    unsigned bits = sel ? o1 : o0;
    float u = __uint_as_float((bits >> 9) | 0x3f800000u) - 1.0f;  // [0,1)
    s = (input[e] > u) ? 1.0f : 0.0f;
    xh[idx] = (_Float16)s;
    p1 = s * A1[k];
    p2 = s * A2[k];
  } else {
    xh[idx] = (_Float16)0.0f;   // K padding stays zero every step
  }
  block_reduce3_atomic(p1, p2, s, red, acc);
}

// ---------------- fused GEMM1 + LIF1: h1m/h1s update in WMMA epilogue -----------
// pre = x @ W1^T + b1 ; m = m*DECAY*(1-s_prev) + pre ; s = m > THRESH
// Wave computes a 32x64 tile = 2x4 16x16 WMMA tiles; K loop over 800 in steps of 32.
__global__ __launch_bounds__(256) void snn_gemm1_lif1(const _Float16* __restrict__ xh,
                                                      const _Float16* __restrict__ w1h,
                                                      const float* __restrict__ b1,
                                                      float* __restrict__ h1m,
                                                      _Float16* __restrict__ h1s,
                                                      const float* __restrict__ A1,
                                                      const float* __restrict__ A2,
                                                      float* __restrict__ acc_g) {
  __shared__ float red[256];
  const int lane = threadIdx.x & 31;
  const int wave = threadIdx.x >> 5;
  const int w = blockIdx.x * 8 + wave;   // 4096 waves total
  const int m0 = (w >> 4) * 32;          // 256 M-strips
  const int n0 = (w & 15) * 64;          // 16 N-strips
  const int r = lane & 15;
  const int half = lane >> 4;

  v8f acc[2][4] = {};
  for (int k0 = 0; k0 < KP_; k0 += 32) {
    AFrag a[2], bq[4];
    for (int mt = 0; mt < 2; ++mt) {
      const _Float16* p = xh + (long)(m0 + mt * 16 + r) * KP_ + k0 + half * 8;
      a[mt].h[0] = *(const v8h*)(p);
      a[mt].h[1] = *(const v8h*)(p + 16);
    }
    for (int nt = 0; nt < 4; ++nt) {
      const _Float16* p = w1h + (long)(n0 + nt * 16 + r) * KP_ + k0 + half * 8;
      bq[nt].h[0] = *(const v8h*)(p);
      bq[nt].h[1] = *(const v8h*)(p + 16);
    }
    for (int mt = 0; mt < 2; ++mt)
      for (int nt = 0; nt < 4; ++nt)
        acc[mt][nt] = __builtin_amdgcn_wmma_f32_16x16x32_f16(
            false, a[mt].v, false, bq[nt].v, (short)0, acc[mt][nt], false, false);
  }

  // LIF epilogue: each output element is owned by exactly one lane.
  float p1 = 0.0f, p2 = 0.0f, p3 = 0.0f;
  for (int nt = 0; nt < 4; ++nt) {
    const int n = n0 + nt * 16 + r;
    const float bias = b1[n];
    const float a1n = A1[IN_ + n];
    const float a2n = A2[IN_ + n];
    for (int mt = 0; mt < 2; ++mt) {
      for (int j = 0; j < 8; ++j) {
        const int row = m0 + mt * 16 + j + 8 * half;
        const long off = (long)row * H_ + n;
        const float pre = acc[mt][nt][j] + bias;
        float m = h1m[off];
        float sp = (float)h1s[off];
        m = m * DECAY * (1.0f - sp) + pre;
        float s = (m > THRESH) ? 1.0f : 0.0f;
        h1m[off] = m;
        h1s[off] = (_Float16)s;
        p1 += s * a1n;
        p2 += s * a2n;
        p3 += s;
      }
    }
  }
  block_reduce3_atomic(p1, p2, p3, red, acc_g);
}

// ---------------- fused GEMM2 + LIF2: h2m/h2s/h2sum update in epilogue ----------
__global__ __launch_bounds__(256) void snn_gemm2_lif2(const _Float16* __restrict__ h1s,
                                                      const _Float16* __restrict__ w2h,
                                                      const float* __restrict__ b2,
                                                      float* __restrict__ h2m,
                                                      float* __restrict__ h2s,
                                                      float* __restrict__ h2sum) {
  const int lane = threadIdx.x & 31;
  const int wave = threadIdx.x >> 5;
  const int w = blockIdx.x * 8 + wave;   // 512 waves, one 16x16 tile each
  const int m0 = w * 16;
  const int r = lane & 15;
  const int half = lane >> 4;

  v8f acc = {};
  for (int k0 = 0; k0 < H_; k0 += 32) {
    AFrag a, bq;
    const _Float16* pa = h1s + (long)(m0 + r) * H_ + k0 + half * 8;
    a.h[0] = *(const v8h*)(pa);
    a.h[1] = *(const v8h*)(pa + 16);
    const _Float16* pb = w2h + (long)r * H_ + k0 + half * 8;
    bq.h[0] = *(const v8h*)(pb);
    bq.h[1] = *(const v8h*)(pb + 16);
    acc = __builtin_amdgcn_wmma_f32_16x16x32_f16(
        false, a.v, false, bq.v, (short)0, acc, false, false);
  }
  const float bias = (r < OUT_) ? b2[r] : 0.0f;
  for (int j = 0; j < 8; ++j) {
    const int row = m0 + j + 8 * half;
    const long off = (long)row * N2P_ + r;
    const float pre = acc[j] + bias;
    float m = h2m[off];
    float sp = h2s[off];
    m = m * DECAY * (1.0f - sp) + pre;
    float s = (m > THRESH) ? 1.0f : 0.0f;
    h2m[off] = m;
    h2s[off] = s;
    h2sum[off] += s;
  }
}

// ---------------- final packing: outputs + scalars ----------------
__global__ void snn_final(const float* __restrict__ h2sum,
                          const float* __restrict__ acc,
                          float* __restrict__ out) {
  const long idx = (long)blockIdx.x * blockDim.x + threadIdx.x;
  if (idx < (long)B_ * OUT_) {
    const int b = (int)(idx / OUT_), o = (int)(idx % OUT_);
    out[idx] = h2sum[(long)b * N2P_ + o] * (1.0f / (float)TW_);
  }
  if (idx < 3) out[(long)B_ * OUT_ + idx] = acc[idx];
}

// ---------------- host entry ----------------
extern "C" void kernel_launch(void* const* d_in, const int* in_sizes, int n_in,
                              void* d_out, int out_size, void* d_ws, size_t ws_size,
                              hipStream_t stream) {
  (void)in_sizes; (void)n_in; (void)out_size; (void)ws_size;
  const float* input = (const float*)d_in[0];
  const float* W1    = (const float*)d_in[1];
  const float* b1    = (const float*)d_in[2];
  const float* W2    = (const float*)d_in[3];
  const float* b2    = (const float*)d_in[4];
  const float* vals  = (const float*)d_in[5];
  const int*   cnts  = (const int*)d_in[6];
  float* out = (float*)d_out;

  char* ws = (char*)d_ws;
  size_t off = 0;
  auto alloc = [&](size_t bytes) -> char* {
    char* p = ws + off;
    off = (off + bytes + 255) & ~(size_t)255;
    return p;
  };
  _Float16* xh   = (_Float16*)alloc((size_t)B_ * KP_ * 2);
  _Float16* w1h  = (_Float16*)alloc((size_t)H_ * KP_ * 2);
  _Float16* w2h  = (_Float16*)alloc((size_t)N2P_ * H_ * 2);
  float*    h1m  = (float*)alloc((size_t)B_ * H_ * 4);
  _Float16* h1s  = (_Float16*)alloc((size_t)B_ * H_ * 2);
  float*    h2m  = (float*)alloc((size_t)B_ * N2P_ * 4);
  float*    h2s  = (float*)alloc((size_t)B_ * N2P_ * 4);
  float*    h2sum= (float*)alloc((size_t)B_ * N2P_ * 4);
  float*    A1   = (float*)alloc((size_t)TABLE_ * 4);
  float*    A2   = (float*)alloc((size_t)TABLE_ * 4);
  float*    acc  = (float*)alloc(4 * 4);

  // Re-initialize all persistent state in-graph (stateless across replays).
  snn_zero<<<2048, 256, 0, stream>>>(h1m, (long)B_ * H_);
  snn_zero<<<2048, 256, 0, stream>>>((float*)h1s, (long)B_ * H_ / 2);
  snn_zero<<<512, 256, 0, stream>>>(h2m, (long)B_ * N2P_);
  snn_zero<<<512, 256, 0, stream>>>(h2s, (long)B_ * N2P_);
  snn_zero<<<512, 256, 0, stream>>>(h2sum, (long)B_ * N2P_);
  snn_zero<<<1, 32, 0, stream>>>(acc, 4);

  snn_tables<<<TABLE_, 256, 0, stream>>>(vals, cnts, A1, A2);
  snn_pack_w1<<<(H_ * KP_ + 255) / 256, 256, 0, stream>>>(W1, w1h);
  snn_pack_w2<<<(N2P_ * H_ + 255) / 256, 256, 0, stream>>>(W2, w2h);

  // jax.random.split(jax.random.key(42), 20) on host:
  // flat = threefry2x32((0,42), iota(40)); keys[j] = (flat[2j], flat[2j+1])
  unsigned flat[2 * TW_];
  for (int i = 0; i < TW_; ++i) {
    unsigned o0, o1;
    threefry2x32(0u, 42u, (unsigned)i, (unsigned)(i + TW_), o0, o1);
    flat[i] = o0; flat[TW_ + i] = o1;
  }

  for (int t = 0; t < TW_; ++t) {
    const unsigned sk0 = flat[2 * t], sk1 = flat[2 * t + 1];
    snn_spikes<<<(B_ * KP_) / 256, 256, 0, stream>>>(input, xh, A1, A2, acc, sk0, sk1);
    snn_gemm1_lif1<<<512, 256, 0, stream>>>(xh, w1h, b1, h1m, h1s, A1, A2, acc);
    snn_gemm2_lif2<<<64, 256, 0, stream>>>(h1s, w2h, b2, h2m, h2s, h2sum);
  }
  snn_final<<<(B_ * OUT_ + 255) / 256, 256, 0, stream>>>(h2sum, acc, out);
}
